// CDBNE_77541339562095
// MI455X (gfx1250) — matrix operand order repaired and checked by reading, not compiled, not run on hardware
//
#include <hip/hip_runtime.h>
#include <hip/hip_bf16.h>
#include <math.h>

#define IN_DIM     128
#define HID_DIM    128
#define EMB_DIM    64
#define N_CLUST    32
#define NEG_SLOPE  0.2f

typedef __attribute__((ext_vector_type(16))) __bf16 v16bf;
typedef __attribute__((ext_vector_type(8)))  float  v8f;

struct alignas(16) U4 { unsigned a, b, c, d; };
struct alignas(16) F4 { float x, y, z, w; };
struct alignas(8)  H4 { unsigned short a, b, c, d; };
union Frag16 {
    v16bf v;
    unsigned short s[16];
    U4 q[2];
};

__device__ __forceinline__ float fatomic_add(float* p, float v) {
    return __hip_atomic_fetch_add(p, v, __ATOMIC_RELAXED, __HIP_MEMORY_SCOPE_AGENT);
}
// order-preserving encode for float atomicMax on unsigned
__device__ __forceinline__ unsigned fenc(float f) {
    unsigned u = __float_as_uint(f);
    return (u & 0x80000000u) ? ~u : (u | 0x80000000u);
}
__device__ __forceinline__ float fdec(unsigned u) {
    return (u & 0x80000000u) ? __uint_as_float(u & 0x7FFFFFFFu) : __uint_as_float(~u);
}
__device__ __forceinline__ float leaky(float x) { return x >= 0.f ? x : NEG_SLOPE * x; }
__device__ __forceinline__ unsigned short bf16rne(float f) {
    unsigned u = __float_as_uint(f);
    return (unsigned short)((u + 0x7FFFu + ((u >> 16) & 1u)) >> 16);
}

// ---------------- conversions / fills ----------------
__global__ void k_f32_to_bf16_v4(const float* __restrict__ in, unsigned short* __restrict__ out, int n4) {
    int i = blockIdx.x * blockDim.x + threadIdx.x;
    if (i >= n4) return;
    F4 f = ((const F4*)in)[i];
    H4 h;
    h.a = bf16rne(f.x); h.b = bf16rne(f.y); h.c = bf16rne(f.z); h.d = bf16rne(f.w);
    ((H4*)out)[i] = h;
}
__global__ void k_fill_u32(unsigned* p, unsigned v, int n) {
    int i = blockIdx.x * blockDim.x + threadIdx.x;
    if (i < n) p[i] = v;
}
__global__ void k_fill_f32(float* p, float v, int n) {
    int i = blockIdx.x * blockDim.x + threadIdx.x;
    if (i < n) p[i] = v;
}
// init accumulator with bias row: out[n*D + d] = b[d]
__global__ void k_init_bias(float* __restrict__ out, const float* __restrict__ b, int total, int Dmask) {
    int t = blockIdx.x * blockDim.x + threadIdx.x;
    if (t < total) out[t] = b[t & Dmask];
}
__global__ void k_relu(float* __restrict__ out, int total) {
    int t = blockIdx.x * blockDim.x + threadIdx.x;
    if (t < total) { float v = out[t]; out[t] = v > 0.f ? v : 0.f; }
}

// ---------------- pack weights (f32 -> bf16, WMMA B-fragment order) ----------------
// Bp[((tn*(K/32)+ks)*32 + lane)*16 + e] = bf16( W[(ks*32 + (lane>>4)*8 + (e&7) + (e>>3)*16)*Nn + tn*16 + (lane&15)] )
__global__ void k_pack_w_bf16(const float* __restrict__ W, unsigned short* __restrict__ Bp,
                              int Nn, int K) {
    int t = blockIdx.x * blockDim.x + threadIdx.x;
    if (t >= Nn * K) return;
    int e    = t & 15;
    int lane = (t >> 4) & 31;
    int ks   = (t >> 9) % (K >> 5);
    int tn   = (t >> 9) / (K >> 5);
    int k = ks * 32 + (lane >> 4) * 8 + (e & 7) + (e >> 3) * 16;
    int n = tn * 16 + (lane & 15);
    Bp[t] = bf16rne(W[(size_t)k * Nn + n]);
}

// ---------------- WMMA GEMM: C[M,NN] = A[M,K](bf16) * B(packed bf16) ----------------
// 4 waves per block, one 16x16 C tile per wave; K and NN compile-time
template <int K, int NN>
__global__ void k_gemm_bf16_wmma(const unsigned short* __restrict__ A,
                                 const unsigned short* __restrict__ Bp,
                                 float* __restrict__ C, int M) {
    constexpr int ntiles = NN >> 4;
    const int mtiles = (M + 15) >> 4;
    const int tile = blockIdx.x * 4 + (threadIdx.x >> 5);
    if (tile >= mtiles * ntiles) return;          // wave-uniform exit
    const int lane = threadIdx.x & 31;
    const int half = lane >> 4;
    const int l15  = lane & 15;
    const int tm = tile / ntiles, tn = tile % ntiles;
    const int m0 = tm << 4, n0 = tn << 4;
    const int row = m0 + l15;
    const int rr  = row < M ? row : (M - 1);      // clamped rows never stored

    const unsigned short* ab = A + (size_t)rr * K + half * 8;
    const U4* bb = (const U4*)Bp + (size_t)tn * (K / 32) * 64 + lane * 2;

    v8f acc = {};
#pragma unroll
    for (int ks = 0; ks < K / 32; ++ks) {
        Frag16 fa, fb;
        fa.q[0] = *(const U4*)(ab + ks * 32);        // K = k0+half*8 .. +7
        fa.q[1] = *(const U4*)(ab + ks * 32 + 16);   // K = k0+16+half*8 .. +7
        fb.q[0] = bb[ks * 64];                       // packed: 32B contiguous per lane
        fb.q[1] = bb[ks * 64 + 1];
        acc = __builtin_amdgcn_wmma_f32_16x16x32_bf16(false, fa.v, false, fb.v,
                                                      (short)0, acc, false, false);
    }
    float* cp = C + (size_t)(m0 + half * 8) * NN + n0 + l15;
    if (m0 + 16 <= M) {                           // full tile (always true for M%16==0)
#pragma unroll
        for (int r = 0; r < 8; ++r) cp[r * NN] = acc[r];   // immediate offsets
    } else {
#pragma unroll
        for (int r = 0; r < 8; ++r)
            if (m0 + half * 8 + r < M) cp[r * NN] = acc[r];
    }
}

// ---------------- attention scores: es = h.a_s, ed = h.a_d ----------------
__global__ void k_scores(const float* __restrict__ h, const float* __restrict__ as_,
                         const float* __restrict__ ad_, float* __restrict__ es,
                         float* __restrict__ ed, int N, int D) {
    int n = blockIdx.x * blockDim.x + threadIdx.x;
    if (n >= N) return;
    const F4* hp = (const F4*)(h + (size_t)n * D);
    const F4* sp = (const F4*)as_;
    const F4* dp = (const F4*)ad_;
    float s = 0.f, d = 0.f;
    for (int i = 0; i < (D >> 2); ++i) {
        F4 hv = hp[i], sv = sp[i], dv = dp[i];
        s += hv.x * sv.x + hv.y * sv.y + hv.z * sv.z + hv.w * sv.w;
        d += hv.x * dv.x + hv.y * dv.y + hv.z * dv.z + hv.w * dv.w;
    }
    es[n] = s; ed[n] = d;
}

// ---------------- edge softmax passes ----------------
__global__ void k_edge_max(const int* __restrict__ ei, const float* __restrict__ es,
                           const float* __restrict__ ed, unsigned* __restrict__ emax,
                           int E, int Etot) {
    int e = blockIdx.x * blockDim.x + threadIdx.x;
    if (e >= Etot) return;
    int s, d;
    if (e < E) { s = ei[e]; d = ei[E + e]; } else { s = d = e - E; }  // self loops
    float v = leaky(es[s] + ed[d]);
    atomicMax(&emax[d], fenc(v));
}

__global__ void k_edge_expsum(const int* __restrict__ ei, const float* __restrict__ es,
                              const float* __restrict__ ed, const unsigned* __restrict__ emax,
                              float* __restrict__ exv, float* __restrict__ denom,
                              int E, int Etot) {
    int e = blockIdx.x * blockDim.x + threadIdx.x;
    if (e >= Etot) return;
    int s, d;
    if (e < E) { s = ei[e]; d = ei[E + e]; } else { s = d = e - E; }
    float m = fdec(emax[d]);
    float w = expf(leaky(es[s] + ed[d]) - m);
    exv[e] = w;
    fatomic_add(&denom[d], w);
}

// one wave per edge; lanes stride the feature dim
__global__ void k_edge_agg(const int* __restrict__ ei, const float* __restrict__ exv,
                           const float* __restrict__ denom, const float* __restrict__ h,
                           float* __restrict__ out, int E, int Etot, int D) {
    int t = blockIdx.x * blockDim.x + threadIdx.x;
    int e = t >> 5, lane = t & 31;
    if (e >= Etot) return;
    int s, d;
    if (e < E) { s = ei[e]; d = ei[E + e]; } else { s = d = e - E; }
    float alpha = exv[e] / denom[d];
    const float* hs = h + (size_t)s * D;
    float* od = out + (size_t)d * D;
    for (int c = lane; c < D; c += 32) fatomic_add(&od[c], alpha * hs[c]);
}

// ---------------- Student-t cluster assignment: one wave per node, lane = cluster ----------------
__global__ void k_cluster_q(const float* __restrict__ z, const float* __restrict__ clus,
                            float* __restrict__ qout, int N) {
    int gid = blockIdx.x * blockDim.x + threadIdx.x;
    int node = gid >> 5, lane = gid & 31;
    if (node >= N) return;                       // whole wave exits together
    const float* zp = z + (size_t)node * EMB_DIM;
    const float* cp = clus + (size_t)lane * EMB_DIM;
    float d2 = 0.f;
#pragma unroll 8
    for (int i = 0; i < EMB_DIM; ++i) { float df = zp[i] - cp[i]; d2 += df * df; }
    float q = rsqrtf(1.0f + d2);
    float tot = q;
    for (int o = 16; o > 0; o >>= 1) tot += __shfl_xor(tot, o, 32);
    qout[(size_t)node * N_CLUST + lane] = q / tot;
}

// ---------------- host ----------------
static inline int ceildiv(int a, int b) { return (a + b - 1) / b; }

extern "C" void kernel_launch(void* const* d_in, const int* in_sizes, int n_in,
                              void* d_out, int out_size, void* d_ws, size_t ws_size,
                              hipStream_t stream) {
    (void)n_in; (void)out_size; (void)ws_size;
    const float* x    = (const float*)d_in[0];
    const int*   ei   = (const int*)d_in[1];
    const float* clus = (const float*)d_in[2];
    const float* Wm[4] = {(const float*)d_in[3], (const float*)d_in[7],
                          (const float*)d_in[11], (const float*)d_in[15]};
    const float* As[4] = {(const float*)d_in[4], (const float*)d_in[8],
                          (const float*)d_in[12], (const float*)d_in[16]};
    const float* Ad[4] = {(const float*)d_in[5], (const float*)d_in[9],
                          (const float*)d_in[13], (const float*)d_in[17]};
    const float* Bb[4] = {(const float*)d_in[6], (const float*)d_in[10],
                          (const float*)d_in[14], (const float*)d_in[18]};

    const int N    = in_sizes[0] / IN_DIM;   // 50000
    const int E    = in_sizes[1] / 2;        // 800000
    const int Etot = E + N;                  // + self loops

    // workspace carve-out (≈94 MB)
    char* wp = (char*)d_ws;
    auto take = [&](size_t bytes) -> char* {
        char* r = wp; wp += (bytes + 255) & ~(size_t)255; return r;
    };
    unsigned short* bfA = (unsigned short*)take((size_t)N * 128 * 2);
    unsigned short* bfW = (unsigned short*)take(128 * 128 * 2);   // packed B fragments
    float*    hbuf  = (float*)take((size_t)N * 128 * 4);
    float*    h1    = (float*)take((size_t)N * 128 * 4);
    float*    h2    = (float*)take((size_t)N * 128 * 4);
    float*    exv   = (float*)take((size_t)Etot * 4);
    unsigned* emax  = (unsigned*)take((size_t)N * 4);
    float*    denom = (float*)take((size_t)N * 4);
    float*    es    = (float*)take((size_t)N * 4);
    float*    ed    = (float*)take((size_t)N * 4);

    float* z_out    = (float*)d_out;                 // [N,64]
    float* xhat_out = z_out + (size_t)N * EMB_DIM;   // [N,128]
    float* q_out    = xhat_out + (size_t)N * IN_DIM; // [N,32]

    auto run_layer = [&](const float* in, int din, int dout, const float* Wp,
                         const float* asp, const float* adp, const float* bp,
                         float* outp, int relu) {
        int nIn = N * din, nW = din * dout, nOut = N * dout;
        k_f32_to_bf16_v4<<<ceildiv(nIn / 4, 256), 256, 0, stream>>>(in, bfA, nIn / 4);
        k_pack_w_bf16<<<ceildiv(nW, 256), 256, 0, stream>>>(Wp, bfW, dout, din);
        int tiles = ceildiv(N, 16) * (dout / 16);
        if (din == 128 && dout == 128)
            k_gemm_bf16_wmma<128, 128><<<ceildiv(tiles, 4), 128, 0, stream>>>(bfA, bfW, hbuf, N);
        else if (din == 128)
            k_gemm_bf16_wmma<128, 64><<<ceildiv(tiles, 4), 128, 0, stream>>>(bfA, bfW, hbuf, N);
        else
            k_gemm_bf16_wmma<64, 128><<<ceildiv(tiles, 4), 128, 0, stream>>>(bfA, bfW, hbuf, N);
        k_scores<<<ceildiv(N, 256), 256, 0, stream>>>(hbuf, asp, adp, es, ed, N, dout);
        k_fill_u32<<<ceildiv(N, 256), 256, 0, stream>>>(emax, 0x007FFFFFu, N);  // enc(-inf)
        k_fill_f32<<<ceildiv(N, 256), 256, 0, stream>>>(denom, 0.f, N);
        k_init_bias<<<ceildiv(nOut, 256), 256, 0, stream>>>(outp, bp, nOut, dout - 1);
        k_edge_max<<<ceildiv(Etot, 256), 256, 0, stream>>>(ei, es, ed, emax, E, Etot);
        k_edge_expsum<<<ceildiv(Etot, 256), 256, 0, stream>>>(ei, es, ed, emax, exv, denom, E, Etot);
        k_edge_agg<<<ceildiv(Etot * 32, 256), 256, 0, stream>>>(ei, exv, denom, hbuf, outp, E, Etot, dout);
        if (relu) k_relu<<<ceildiv(nOut, 256), 256, 0, stream>>>(outp, nOut);
    };

    run_layer(x,     IN_DIM,  HID_DIM, Wm[0], As[0], Ad[0], Bb[0], h1,       1);
    run_layer(h1,    HID_DIM, EMB_DIM, Wm[1], As[1], Ad[1], Bb[1], z_out,    0);
    run_layer(z_out, EMB_DIM, HID_DIM, Wm[2], As[2], Ad[2], Bb[2], h2,       1);
    run_layer(h2,    HID_DIM, IN_DIM,  Wm[3], As[3], Ad[3], Bb[3], xhat_out, 0);

    k_cluster_q<<<ceildiv(N * N_CLUST, 256), 256, 0, stream>>>(z_out, clus, q_out, N);
}
